// GraphMETNetwork_dyn_40063454937529
// MI455X (gfx1250) — matrix operand order repaired and checked by reading, not compiled, not run on hardware
//
#include <hip/hip_runtime.h>
#include <hip/hip_bf16.h>

// ---------------------------------------------------------------------------
// GraphMET network on gfx1250 (CDNA5). All GEMMs via v_wmma_f32_16x16x32_bf16.
// Sizes: B=32 events, N=2048 nodes/event, H=128, K=16, n=B*N=65536.
// ---------------------------------------------------------------------------

#define NN_TOTAL 65536   // B*N
#define NEV      32
#define NPEV     2048
#define HDIM     128
#define KNN      16

typedef __attribute__((ext_vector_type(16))) __bf16 v16bf;
typedef __attribute__((ext_vector_type(8)))  float  v8f;

union Frag { v16bf v; unsigned u[8]; };

__device__ inline unsigned short f2bf(float x) {
  unsigned u = __float_as_uint(x);
  u += 0x7FFFu + ((u >> 16) & 1u);          // round-to-nearest-even
  return (unsigned short)(u >> 16);
}
__device__ inline float bf2f(unsigned short h) {
  return __uint_as_float(((unsigned)h) << 16);
}
__device__ inline float elu(float x) { return x > 0.f ? x : (__expf(x) - 1.f); }

// A-matrix 16x32 bf16 fragment: lanes 0-15 hold K=0..15, lanes 16-31 K=16..31
// (2 bf16 per dword, contiguous K, matching load_b_frag). Row = lane&15.
__device__ inline void load_a_frag(const unsigned short* p0, int ld, int row0,
                                   int kb, int lane, Frag& f) {
  int half = (lane >> 4) & 1, r = lane & 15;
  const unsigned short* p = p0 + (size_t)(row0 + r) * ld + kb;
#pragma unroll
  for (int i = 0; i < 4; ++i) f.u[i]     = *(const unsigned*)(p + half * 16 + 2 * i);
#pragma unroll
  for (int i = 0; i < 4; ++i) f.u[4 + i] = *(const unsigned*)(p + 8 + half * 16 + 2 * i);
}

// B-matrix 32x16 bf16 fragment: lanes 0-15 hold K=0..15, lanes 16-31 K=16..31,
// 2 bf16 per VGPR, column N = lane&15. We feed rows of a row-major [out,K]
// matrix (i.e. B[k][n] = W[n][k]) so both X@W^T and X@X^T use this loader.
__device__ inline void load_b_frag(const unsigned short* p0, int ld, int row0,
                                   int kb, int lane, Frag& f) {
  int half = (lane >> 4) & 1, r = lane & 15;
  const unsigned short* p = p0 + (size_t)(row0 + r) * ld + kb + half * 16;
#pragma unroll
  for (int i = 0; i < 8; ++i) f.u[i] = *(const unsigned*)(p + 2 * i);
}

// ---------------------------------------------------------------------------
// 1) Per-node feature MLP: cont (7->64) + categorical lookups (96->64), ELU,
//    writes bf16 features [emb_cat | emb_cont] for the encoder WMMA GEMM.
// ---------------------------------------------------------------------------
__global__ __launch_bounds__(256) void feat_kernel(
    const float* __restrict__ x, const float* __restrict__ emb_charge,
    const float* __restrict__ emb_pdgid, const float* __restrict__ emb_pv,
    const float* __restrict__ W_cont, const float* __restrict__ b_cont,
    const float* __restrict__ W_cat, const float* __restrict__ b_cat,
    unsigned short* __restrict__ Xbf) {
  int i = blockIdx.x * blockDim.x + threadIdx.x;
  if (i >= NN_TOTAL) return;
  const float* xr = x + (size_t)i * 11;

  float xc[7];
#pragma unroll
  for (int c = 0; c < 7; ++c) xc[c] = xr[c];

  int chrg = (int)xr[9] + 1;                       // 0..2
  int pv = (int)xr[10];                            // 0..3
  int praw = (int)xr[8]; if (praw < 0) praw = -praw;
  const int PD0=1,PD1=2,PD2=11,PD3=13,PD4=22,PD5=130,PD6=211;
  int pidx = 0;
  pidx = (praw==PD0)?0:(praw==PD1)?1:(praw==PD2)?2:(praw==PD3)?3:
         (praw==PD4)?4:(praw==PD5)?5:(praw==PD6)?6:praw;
  const float* ec = emb_charge + chrg * 32;
  const float* ep = emb_pdgid + pidx * 32;
  const float* ev = emb_pv + pv * 32;

  unsigned short* out = Xbf + (size_t)i * HDIM;
  // emb_cat -> channels [0,64)
#pragma unroll 2
  for (int o = 0; o < 64; ++o) {
    const float* w = W_cat + (size_t)o * 96;
    float a = b_cat[o];
    for (int c = 0; c < 32; ++c) a += ec[c] * w[c];
    for (int c = 0; c < 32; ++c) a += ep[c] * w[32 + c];
    for (int c = 0; c < 32; ++c) a += ev[c] * w[64 + c];
    out[o] = f2bf(elu(a));
  }
  // emb_cont -> channels [64,128)
#pragma unroll 4
  for (int o = 0; o < 64; ++o) {
    const float* w = W_cont + (size_t)o * 7;
    float a = b_cont[o];
#pragma unroll
    for (int c = 0; c < 7; ++c) a += xc[c] * w[c];
    out[64 + o] = f2bf(elu(a));
  }
}

// ---------------------------------------------------------------------------
// 2) Weight prep: bf16 W_enc; EdgeConv decomposition W2 and (W1-W2) in bf16.
// ---------------------------------------------------------------------------
__global__ void prep_w_kernel(const float* __restrict__ Wenc,
                              const float* __restrict__ Wc0,
                              const float* __restrict__ Wc1,
                              unsigned short* WencBf,
                              unsigned short* W2_0, unsigned short* W12_0,
                              unsigned short* W2_1, unsigned short* W12_1) {
  int t = blockIdx.x * blockDim.x + threadIdx.x;
  if (t >= HDIM * HDIM) return;
  int o = t >> 7, k = t & 127;
  WencBf[t] = f2bf(Wenc[t]);
  float w1 = Wc0[(size_t)o * 256 + k], w2 = Wc0[(size_t)o * 256 + 128 + k];
  W2_0[t] = f2bf(w2); W12_0[t] = f2bf(w1 - w2);
  w1 = Wc1[(size_t)o * 256 + k]; w2 = Wc1[(size_t)o * 256 + 128 + k];
  W2_1[t] = f2bf(w2); W12_1[t] = f2bf(w1 - w2);
}

// ---------------------------------------------------------------------------
// 3) C[n,128] = act(X[n,128]_bf16 @ W[128,128]^T_bf16 + bias) via WMMA.
//    8 waves/block, 1 wave per 16-row tile, K=128 in 4 steps of 32.
// ---------------------------------------------------------------------------
__global__ __launch_bounds__(256) void gemm128_kernel(
    const unsigned short* __restrict__ X, const unsigned short* __restrict__ W,
    const float* __restrict__ bias, float* __restrict__ C, int act) {
  int wave = threadIdx.x >> 5, lane = threadIdx.x & 31;
  int mbase = (blockIdx.x * 8 + wave) * 16;
  v8f acc[8];
#pragma unroll
  for (int ct = 0; ct < 8; ++ct) acc[ct] = {};
#pragma unroll
  for (int kt = 0; kt < 4; ++kt) {
    Frag a; load_a_frag(X, HDIM, mbase, kt * 32, lane, a);
#pragma unroll
    for (int ct = 0; ct < 8; ++ct) {
      Frag b; load_b_frag(W, HDIM, ct * 16, kt * 32, lane, b);
      acc[ct] = __builtin_amdgcn_wmma_f32_16x16x32_bf16(
          false, a.v, false, b.v, (short)0, acc[ct], false, false);
    }
  }
  int half = lane >> 4, r = lane & 15;
#pragma unroll
  for (int ct = 0; ct < 8; ++ct) {
    int col = ct * 16 + r;
    float bv = bias ? bias[col] : 0.f;
#pragma unroll
    for (int v = 0; v < 8; ++v) {
      float val = acc[ct][v] + bv;
      if (act) val = elu(val);
      C[(size_t)(mbase + v + half * 8) * HDIM + col] = val;
    }
  }
}

// ---------------------------------------------------------------------------
// 4) BatchNorm helpers: zero stats / per-channel sum,sumsq / finalize / apply.
// ---------------------------------------------------------------------------
__global__ void zero_stats_kernel(float* stats) {
  if (threadIdx.x < 512) stats[threadIdx.x] = 0.f;
}

__global__ __launch_bounds__(128) void stats_kernel(const float* __restrict__ C,
                                                    float* __restrict__ stats) {
  int t = threadIdx.x;                 // channel
  size_t rbase = (size_t)blockIdx.x * 512;
  float s = 0.f, s2 = 0.f;
  for (int i = 0; i < 512; ++i) {
    float v = C[(rbase + i) * HDIM + t];
    s += v; s2 += v * v;
  }
  atomicAdd(&stats[t], s);
  atomicAdd(&stats[128 + t], s2);
}

__global__ void bn_finalize_kernel(float* stats, const float* __restrict__ g,
                                   const float* __restrict__ be) {
  int t = threadIdx.x;
  const float inv_n = 1.f / (float)NN_TOTAL;
  float mean = stats[t] * inv_n;
  float var = stats[128 + t] * inv_n - mean * mean;
  float sc = g[t] * rsqrtf(var + 1e-5f);
  stats[256 + t] = sc;
  stats[384 + t] = be[t] - mean * sc;
}

// normalize, emit bf16 activation + per-node squared norm (for distances)
__global__ __launch_bounds__(256) void bn_apply_kernel(
    const float* __restrict__ C, const float* __restrict__ stats,
    unsigned short* __restrict__ Xbf, float* __restrict__ sq) {
  int node = blockIdx.x * 8 + (threadIdx.x >> 5);
  int lane = threadIdx.x & 31;
  float4 v = ((const float4*)(C + (size_t)node * HDIM))[lane];
  float4 sc = ((const float4*)(stats + 256))[lane];
  float4 sh = ((const float4*)(stats + 384))[lane];
  float x0 = v.x * sc.x + sh.x, x1 = v.y * sc.y + sh.y;
  float x2 = v.z * sc.z + sh.z, x3 = v.w * sc.w + sh.w;
  uint2 u;
  u.x = (unsigned)f2bf(x0) | ((unsigned)f2bf(x1) << 16);
  u.y = (unsigned)f2bf(x2) | ((unsigned)f2bf(x3) << 16);
  ((uint2*)(Xbf + (size_t)node * HDIM))[lane] = u;
  float s = x0 * x0 + x1 * x1 + x2 * x2 + x3 * x3;
#pragma unroll
  for (int off = 16; off; off >>= 1) s += __shfl_xor(s, off, 32);
  if (lane == 0) sq[node] = s;
}

// ---------------------------------------------------------------------------
// 5) kNN: per (event, 16-row tile). Gram via WMMA bf16, d2 = |xi|^2+|xj|^2-2G,
//    full 16x2048 distance rows in LDS (128KB of the 320KB WGP LDS), then 16
//    rounds of wave32 argmin with packed (bits,idx) shfl_xor reduction.
// ---------------------------------------------------------------------------
__global__ __launch_bounds__(256) void knn_kernel(
    const unsigned short* __restrict__ Xbf, const float* __restrict__ sq,
    int* __restrict__ knn) {
  __shared__ float dist[16][NPEV];        // 128 KB
  int wave = threadIdx.x >> 5, lane = threadIdx.x & 31;
  int half = lane >> 4, r = lane & 15;
  int bbase = blockIdx.y * NPEV;
  int mbase = bbase + blockIdx.x * 16;

  Frag a[4];
#pragma unroll
  for (int kt = 0; kt < 4; ++kt) load_a_frag(Xbf, HDIM, mbase, kt * 32, lane, a[kt]);
  float sqi[8];
#pragma unroll
  for (int v = 0; v < 8; ++v) sqi[v] = sq[mbase + v + half * 8];

  for (int jt = wave; jt < NPEV / 16; jt += 8) {
    v8f acc = {};
#pragma unroll
    for (int kt = 0; kt < 4; ++kt) {
      Frag b; load_b_frag(Xbf, HDIM, bbase + jt * 16, kt * 32, lane, b);
      acc = __builtin_amdgcn_wmma_f32_16x16x32_bf16(
          false, a[kt].v, false, b.v, (short)0, acc, false, false);
    }
    float sqj = sq[bbase + jt * 16 + r];
#pragma unroll
    for (int v = 0; v < 8; ++v) {
      float d2 = sqi[v] + sqj - 2.f * acc[v];
      dist[v + half * 8][jt * 16 + r] = fmaxf(d2, 0.f);   // keep uint-orderable
    }
  }
  __syncthreads();

  // 8 waves x 2 rows: 16 rounds of argmin over 2048, invalidating the winner.
  for (int rr = 0; rr < 2; ++rr) {
    int row = wave + rr * 8;
    int gi = mbase + row - bbase;                 // row within event (unused)
    (void)gi;
    int* dst = knn + (size_t)(mbase + row) * KNN;
    for (int k = 0; k < KNN; ++k) {
      float best = 3.402823466e38f; int bidx = 0;
#pragma unroll 8
      for (int c = lane; c < NPEV; c += 32) {
        float v = dist[row][c];
        if (v < best) { best = v; bidx = c; }
      }
      unsigned long long pk =
          (((unsigned long long)__float_as_uint(best)) << 32) | (unsigned)bidx;
#pragma unroll
      for (int off = 16; off; off >>= 1) {
        unsigned long long o = __shfl_xor(pk, off, 32);
        if (o < pk) pk = o;
      }
      int minidx = (int)(pk & 0xFFFFFFFFu);
      if (lane == 0) {
        dst[k] = bbase + minidx;                  // global neighbor index
        dist[row][minidx] = 3.402823466e38f;
      }
      asm volatile("s_wait_dscnt 0" ::: "memory"); // order LDS inval in-wave
    }
  }
}

// ---------------------------------------------------------------------------
// 6) EdgeConv epilogue: out[i,:] = A[i,:] + max_k Y[knn[i][k],:]
// ---------------------------------------------------------------------------
__global__ __launch_bounds__(256) void gather_max_kernel(
    const float* __restrict__ Y, const float* __restrict__ A,
    const int* __restrict__ knn, float* __restrict__ C) {
  int node = blockIdx.x * 8 + (threadIdx.x >> 5);
  int lane = threadIdx.x & 31;
  const int* nb = knn + (size_t)node * KNN;
  float4 m = make_float4(-3.402823466e38f, -3.402823466e38f,
                         -3.402823466e38f, -3.402823466e38f);
#pragma unroll
  for (int k = 0; k < KNN; ++k) {
    int j = nb[k];
    float4 v = ((const float4*)(Y + (size_t)j * HDIM))[lane];
    m.x = fmaxf(m.x, v.x); m.y = fmaxf(m.y, v.y);
    m.z = fmaxf(m.z, v.z); m.w = fmaxf(m.w, v.w);
  }
  float4 av = ((const float4*)(A + (size_t)node * HDIM))[lane];
  float4 o = make_float4(av.x + m.x, av.y + m.y, av.z + m.z, av.w + m.w);
  ((float4*)(C + (size_t)node * HDIM))[lane] = o;
}

// ---------------------------------------------------------------------------
// 7) Head: out[i] = ELU(emb @ Wo1^T + bo1) @ Wo2^T + bo2  (one wave / node)
// ---------------------------------------------------------------------------
__global__ __launch_bounds__(256) void head_kernel(
    const unsigned short* __restrict__ Xbf, const float* __restrict__ Wo1,
    const float* __restrict__ bo1, const float* __restrict__ Wo2,
    const float* __restrict__ bo2, float* __restrict__ out) {
  int node = blockIdx.x * 8 + (threadIdx.x >> 5);
  int lane = threadIdx.x & 31;
  int o0 = lane, o1 = lane + 32;
  float a0 = bo1[o0], a1 = bo1[o1];
  const unsigned short* e = Xbf + (size_t)node * HDIM;
  const float* w0 = Wo1 + (size_t)o0 * HDIM;
  const float* w1 = Wo1 + (size_t)o1 * HDIM;
#pragma unroll 8
  for (int c = 0; c < HDIM; ++c) {
    float ev = bf2f(e[c]);
    a0 += ev * w0[c];
    a1 += ev * w1[c];
  }
  float p = elu(a0) * Wo2[o0] + elu(a1) * Wo2[o1];
#pragma unroll
  for (int off = 16; off; off >>= 1) p += __shfl_xor(p, off, 32);
  if (lane == 0) out[node] = p + bo2[0];
}

// ---------------------------------------------------------------------------
// Launch
// ---------------------------------------------------------------------------
extern "C" void kernel_launch(void* const* d_in, const int* in_sizes, int n_in,
                              void* d_out, int out_size, void* d_ws, size_t ws_size,
                              hipStream_t stream) {
  (void)in_sizes; (void)n_in; (void)out_size; (void)ws_size;
  const float* x          = (const float*)d_in[0];
  const float* emb_charge = (const float*)d_in[1];
  const float* emb_pdgid  = (const float*)d_in[2];
  const float* emb_pv     = (const float*)d_in[3];
  const float* W_cont = (const float*)d_in[4];
  const float* b_cont = (const float*)d_in[5];
  const float* W_cat  = (const float*)d_in[6];
  const float* b_cat  = (const float*)d_in[7];
  const float* W_enc  = (const float*)d_in[8];
  const float* b_enc  = (const float*)d_in[9];
  const float* g_all  = (const float*)d_in[10];
  const float* be_all = (const float*)d_in[11];
  const float* W_c0 = (const float*)d_in[12];
  const float* b_c0 = (const float*)d_in[13];
  const float* g_c0 = (const float*)d_in[14];
  const float* be_c0 = (const float*)d_in[15];
  const float* W_c1 = (const float*)d_in[16];
  const float* b_c1 = (const float*)d_in[17];
  const float* g_c1 = (const float*)d_in[18];
  const float* be_c1 = (const float*)d_in[19];
  const float* W_o1 = (const float*)d_in[20];
  const float* b_o1 = (const float*)d_in[21];
  const float* W_o2 = (const float*)d_in[22];
  const float* b_o2 = (const float*)d_in[23];
  float* out = (float*)d_out;

  const size_t nN = NN_TOTAL;
  float* B0 = (float*)d_ws;                                  // n*128 f32
  float* B1 = B0 + nN * HDIM;
  float* B2 = B1 + nN * HDIM;
  unsigned short* Xbf = (unsigned short*)(B2 + nN * HDIM);   // n*128 bf16
  float* sq = (float*)(Xbf + nN * HDIM);                     // n f32
  int* knn = (int*)(sq + nN);                                // n*16 i32
  float* stats = (float*)(knn + nN * KNN);                   // 512 f32
  unsigned short* WencBf = (unsigned short*)(stats + 512);
  unsigned short* W2_0  = WencBf + HDIM * HDIM;
  unsigned short* W12_0 = W2_0 + HDIM * HDIM;
  unsigned short* W2_1  = W12_0 + HDIM * HDIM;
  unsigned short* W12_1 = W2_1 + HDIM * HDIM;

  // Stage 0: features + weight prep
  feat_kernel<<<NN_TOTAL / 256, 256, 0, stream>>>(
      x, emb_charge, emb_pdgid, emb_pv, W_cont, b_cont, W_cat, b_cat, Xbf);
  prep_w_kernel<<<64, 256, 0, stream>>>(W_enc, W_c0, W_c1, WencBf, W2_0, W12_0,
                                        W2_1, W12_1);

  // Stage 1: encoder GEMM (+ELU) -> BN -> bf16 activation + |x|^2
  gemm128_kernel<<<512, 256, 0, stream>>>(Xbf, WencBf, b_enc, B0, 1);
  zero_stats_kernel<<<1, 512, 0, stream>>>(stats);
  stats_kernel<<<128, 128, 0, stream>>>(B0, stats);
  bn_finalize_kernel<<<1, 128, 0, stream>>>(stats, g_all, be_all);
  bn_apply_kernel<<<NN_TOTAL / 8, 256, 0, stream>>>(B0, stats, Xbf, sq);

  // Stage 2: EdgeConv 0
  knn_kernel<<<dim3(NPEV / 16, NEV), 256, 0, stream>>>(Xbf, sq, knn);
  gemm128_kernel<<<512, 256, 0, stream>>>(Xbf, W2_0, nullptr, B0, 0);   // Y
  gemm128_kernel<<<512, 256, 0, stream>>>(Xbf, W12_0, b_c0, B1, 0);     // A
  gather_max_kernel<<<NN_TOTAL / 8, 256, 0, stream>>>(B0, B1, knn, B2);
  zero_stats_kernel<<<1, 512, 0, stream>>>(stats);
  stats_kernel<<<128, 128, 0, stream>>>(B2, stats);
  bn_finalize_kernel<<<1, 128, 0, stream>>>(stats, g_c0, be_c0);
  bn_apply_kernel<<<NN_TOTAL / 8, 256, 0, stream>>>(B2, stats, Xbf, sq);

  // Stage 3: EdgeConv 1
  knn_kernel<<<dim3(NPEV / 16, NEV), 256, 0, stream>>>(Xbf, sq, knn);
  gemm128_kernel<<<512, 256, 0, stream>>>(Xbf, W2_1, nullptr, B0, 0);   // Y
  gemm128_kernel<<<512, 256, 0, stream>>>(Xbf, W12_1, b_c1, B1, 0);     // A
  gather_max_kernel<<<NN_TOTAL / 8, 256, 0, stream>>>(B0, B1, knn, B2);
  zero_stats_kernel<<<1, 512, 0, stream>>>(stats);
  stats_kernel<<<128, 128, 0, stream>>>(B2, stats);
  bn_finalize_kernel<<<1, 128, 0, stream>>>(stats, g_c1, be_c1);
  bn_apply_kernel<<<NN_TOTAL / 8, 256, 0, stream>>>(B2, stats, Xbf, sq);

  // Stage 4: head
  head_kernel<<<NN_TOTAL / 8, 256, 0, stream>>>(Xbf, W_o1, b_o1, W_o2, b_o2, out);
}
